// TransformerDecoderLayer_33947421507650
// MI455X (gfx1250) — matrix operand order repaired and checked by reading, not compiled, not run on hardware
//
#include <hip/hip_runtime.h>

// ---------------- problem constants ----------------
#define BATCH   8
#define SEQ_LQ  1024
#define SEQ_LK  1024
#define DMODEL  512
#define NHEAD   8
#define DHEAD   64
#define USEL    35          // top-k queries (5*ceil(ln 1024))
#define DFF     2048

typedef unsigned short u16;

// ---------------- WMMA helper types ----------------
typedef __attribute__((ext_vector_type(8)))  float  v8f;
typedef __attribute__((ext_vector_type(16))) __bf16 v16bf;
typedef __attribute__((ext_vector_type(8)))  u16    ushort8;

union FragBF {
  v16bf   v;
  u16     s[16];
  ushort8 h[2];
};

__device__ __forceinline__ u16 f2bf(float f) {
  unsigned u = __builtin_bit_cast(unsigned, f);
  u += 0x7FFFu + ((u >> 16) & 1u);          // round-to-nearest-even
  return (u16)(u >> 16);
}

__device__ __forceinline__ v8f v8zero() {
  v8f z;
#pragma unroll
  for (int i = 0; i < 8; i++) z[i] = 0.0f;
  return z;
}

// LDS byte offset of a __shared__ object (low 32 bits of flat address)
#define LDS_OFF(p) ((unsigned)(size_t)(p))

// CDNA5 async global->LDS 128-bit copy (ASYNCcnt-tracked), GVS addressing
__device__ __forceinline__ void async_g2l_b128(unsigned ldsAddr, unsigned gOff,
                                               const void* sBase) {
  asm volatile("global_load_async_to_lds_b128 %0, %1, %2"
               :: "v"(ldsAddr), "v"(gOff), "s"(sBase) : "memory");
}
__device__ __forceinline__ void wait_async0() {
  asm volatile("s_wait_asynccnt 0x0" ::: "memory");
}

// =====================================================================
// bf16 GEMM: C[M,N] = act(A[M,K] @ Wt[N,K]^T + bias[N])
// A, Wt bf16, K-contiguous. Double-buffered async LDS staging: the DMA
// engine fills tile t+1 while WMMAs consume tile t.
// outMode: bit0 = write f32 C, bit1 = write bf16 Cbf.
// =====================================================================
__global__ __launch_bounds__(128)
void gemm_bf16_kernel(const u16* __restrict__ A, const u16* __restrict__ Wt,
                      const float* __restrict__ bias,
                      float* __restrict__ C, u16* __restrict__ Cbf,
                      int M, int N, int K, int act, int outMode)
{
  __shared__ __align__(16) u16 Alds[2][64][40];  // [buf][m][k] 80B rows
  __shared__ __align__(16) u16 Wlds[2][64][40];  // [buf][n][k]

  const int tid   = threadIdx.x;
  const int wave  = tid >> 5;
  const int lane  = tid & 31;
  const int tileM = (int)blockIdx.x * 64;
  const int tileN = (int)blockIdx.y * 64;

  v8f acc[4];
#pragma unroll
  for (int i = 0; i < 4; i++) acc[i] = v8zero();

  const int arow = wave * 16 + (lane & 15);
  const int kb   = (lane >> 4) * 8;    // A frag K sub-base {0,8}
  const int kg   = (lane >> 4) * 8;    // B frag K sub-base {0,8}

  // two 16B chunks per thread per tile: chunk = (row, part)  row<64, part<4
  const int r0c = tid >> 2,          p0 = tid & 3;
  const int r1c = (tid + 128) >> 2,  p1 = tid & 3;
  const unsigned aBase[2] = { LDS_OFF(&Alds[0][0][0]), LDS_OFF(&Alds[1][0][0]) };
  const unsigned wBase[2] = { LDS_OFF(&Wlds[0][0][0]), LDS_OFF(&Wlds[1][0][0]) };

  auto stage = [&](int buf, int k0) {
    async_g2l_b128(aBase[buf] + (unsigned)(r0c * 80 + p0 * 16),
                   (unsigned)(((size_t)(tileM + r0c) * K + k0) * 2) + p0 * 16, A);
    async_g2l_b128(aBase[buf] + (unsigned)(r1c * 80 + p1 * 16),
                   (unsigned)(((size_t)(tileM + r1c) * K + k0) * 2) + p1 * 16, A);
    async_g2l_b128(wBase[buf] + (unsigned)(r0c * 80 + p0 * 16),
                   (unsigned)(((size_t)(tileN + r0c) * K + k0) * 2) + p0 * 16, Wt);
    async_g2l_b128(wBase[buf] + (unsigned)(r1c * 80 + p1 * 16),
                   (unsigned)(((size_t)(tileN + r1c) * K + k0) * 2) + p1 * 16, Wt);
  };

  stage(0, 0);                          // prologue fill
  int buf = 0;
  for (int k0 = 0; k0 < K; k0 += 32, buf ^= 1) {
    wait_async0();                      // current tile landed (overlapped with prev compute)
    __syncthreads();
    if (k0 + 32 < K) stage(buf ^ 1, k0 + 32);   // kick next tile's DMA

    FragBF af;
    af.h[0] = *(const ushort8*)&Alds[buf][arow][kb];
    af.h[1] = *(const ushort8*)&Alds[buf][arow][16 + kb];
#pragma unroll
    for (int nt = 0; nt < 4; nt++) {
      int col = nt * 16 + (lane & 15);
      FragBF bf_;
      bf_.h[0] = *(const ushort8*)&Wlds[buf][col][kg];
      bf_.h[1] = *(const ushort8*)&Wlds[buf][col][kg + 16];
      acc[nt] = __builtin_amdgcn_wmma_f32_16x16x32_bf16(
          false, af.v, false, bf_.v, (short)0, acc[nt], false, false);
    }
  }

  const int rowoff = tileM + wave * 16 + (lane >> 4) * 8;
#pragma unroll
  for (int nt = 0; nt < 4; nt++) {
    int col = tileN + nt * 16 + (lane & 15);
    float bb = bias[col];
#pragma unroll
    for (int r = 0; r < 8; r++) {
      float v = acc[nt][r] + bb;
      if (act) v = fmaxf(v, 0.0f);
      size_t o = (size_t)(rowoff + r) * N + col;
      if (outMode & 1) C[o] = v;
      if (outMode & 2) Cbf[o] = f2bf(v);
    }
  }
}

// =====================================================================
// Fused flash attention on bf16 Q/K/V (f32 accumulate, f32 output).
// 128 threads = 4 waves x 16 query rows; K blocks of 32 keys.
// Double-buffered K (async DMA) and V (vector load + transposed DS store).
// =====================================================================
__global__ __launch_bounds__(128)
void flash_attn_kernel(const u16* __restrict__ Q, int ldq, size_t qBatch, size_t qHead,
                       const u16* __restrict__ Kp, int ldk, size_t kBatch, size_t kHead,
                       const u16* __restrict__ Vp,
                       float* __restrict__ O, int ldo, size_t oBatch, size_t oHead,
                       int qRows, int Lk, float scale, int qTiles)
{
  __shared__ __align__(16) u16 Klds[2][32][72];    // [buf][key][dh] 144B rows
  __shared__ __align__(16) u16 Vtld[2][64][40];    // [buf][dh][key]
  __shared__ __align__(16) u16 Plds[4][16][40];    // per-wave P 16x32

  const int bh = (int)blockIdx.x / qTiles;
  const int qt = (int)blockIdx.x % qTiles;
  const int b  = bh / NHEAD;
  const int h  = bh % NHEAD;

  const u16* q = Q  + (size_t)b * qBatch + (size_t)h * qHead;
  const u16* k = Kp + (size_t)b * kBatch + (size_t)h * kHead;
  const u16* v = Vp + (size_t)b * kBatch + (size_t)h * kHead;
  float*     o = O  + (size_t)b * oBatch + (size_t)h * oHead;

  const int wave = threadIdx.x >> 5;
  const int lane = threadIdx.x & 31;
  const int r0   = qt * 64 + wave * 16;
  const unsigned kBase[2] = { LDS_OFF(&Klds[0][0][0]), LDS_OFF(&Klds[1][0][0]) };

  auto stageK = [&](int buf, int kb0) {
#pragma unroll
    for (int cc = 0; cc < 2; cc++) {
      int c  = threadIdx.x + cc * 128;
      int kk = c >> 3, part = c & 7;
      async_g2l_b128(kBase[buf] + (unsigned)(kk * 144 + part * 16),
                     (unsigned)(((size_t)(kb0 + kk) * ldk) * 2) + part * 16, k);
    }
  };
  auto stageV = [&](int buf, int kb0) {
#pragma unroll
    for (int cc = 0; cc < 2; cc++) {
      int c  = threadIdx.x + cc * 128;
      int kk = c >> 3, db = (c & 7) * 8;
      ushort8 vv = *(const ushort8*)(v + (size_t)(kb0 + kk) * ldk + db);
#pragma unroll
      for (int j = 0; j < 8; j++) Vtld[buf][db + j][kk] = vv[j];
    }
  };

  // ---- Q fragments straight from global bf16 (A-layout) ----
  FragBF qf[2];
  {
    const u16* qp = q + (size_t)(r0 + (lane & 15)) * ldq;
    const int kbq = (lane >> 4) * 8;
#pragma unroll
    for (int kc = 0; kc < 2; kc++) {
      qf[kc].h[0] = *(const ushort8*)(qp + kc * 32 + kbq);
      qf[kc].h[1] = *(const ushort8*)(qp + kc * 32 + 16 + kbq);
    }
  }

  v8f Of[4];
#pragma unroll
  for (int i = 0; i < 4; i++) Of[i] = v8zero();
  float mrun[8], lrun[8];
#pragma unroll
  for (int r = 0; r < 8; r++) { mrun[r] = -3.0e38f; lrun[r] = 0.0f; }

  stageK(0, 0);                        // prologue fill
  stageV(0, 0);
  int buf = 0;
  for (int kb0 = 0; kb0 < Lk; kb0 += 32, buf ^= 1) {
    wait_async0();
    __syncthreads();
    if (kb0 + 32 < Lk) {               // kick next block while computing this one
      stageK(buf ^ 1, kb0 + 32);
      stageV(buf ^ 1, kb0 + 32);
    }

    // ---- S = Q * K^T ----
    v8f S[2];
    S[0] = v8zero(); S[1] = v8zero();
#pragma unroll
    for (int nt = 0; nt < 2; nt++) {
      int key = nt * 16 + (lane & 15);
#pragma unroll
      for (int kc = 0; kc < 2; kc++) {
        int dg = kc * 32 + (lane >> 4) * 8;
        FragBF bf_;
        bf_.h[0] = *(const ushort8*)&Klds[buf][key][dg];
        bf_.h[1] = *(const ushort8*)&Klds[buf][key][dg + 16];
        S[nt] = __builtin_amdgcn_wmma_f32_16x16x32_bf16(
            false, qf[kc].v, false, bf_.v, (short)0, S[nt], false, false);
      }
    }
#pragma unroll
    for (int nt = 0; nt < 2; nt++)
#pragma unroll
      for (int r = 0; r < 8; r++) S[nt][r] *= scale;

    // ---- online softmax (16-lane xor butterflies) ----
    float mb[8];
#pragma unroll
    for (int r = 0; r < 8; r++) mb[r] = fmaxf(S[0][r], S[1][r]);
#pragma unroll
    for (int off = 1; off < 16; off <<= 1)
#pragma unroll
      for (int r = 0; r < 8; r++) mb[r] = fmaxf(mb[r], __shfl_xor(mb[r], off, 32));

    float alpha[8];
#pragma unroll
    for (int r = 0; r < 8; r++) {
      float mn = fmaxf(mrun[r], mb[r]);
      alpha[r] = __expf(mrun[r] - mn);
      mrun[r]  = mn;
    }
    float ls[8];
#pragma unroll
    for (int r = 0; r < 8; r++) ls[r] = 0.0f;
#pragma unroll
    for (int nt = 0; nt < 2; nt++)
#pragma unroll
      for (int r = 0; r < 8; r++) {
        float p = __expf(S[nt][r] - mrun[r]);
        S[nt][r] = p;
        ls[r] += p;
      }
#pragma unroll
    for (int off = 1; off < 16; off <<= 1)
#pragma unroll
      for (int r = 0; r < 8; r++) ls[r] += __shfl_xor(ls[r], off, 32);
#pragma unroll
    for (int r = 0; r < 8; r++) lrun[r] = lrun[r] * alpha[r] + ls[r];
#pragma unroll
    for (int n = 0; n < 4; n++)
#pragma unroll
      for (int r = 0; r < 8; r++) Of[n][r] *= alpha[r];

    // ---- stage P (C-layout -> LDS -> A-layout), wave-private ----
    {
      int prow = (lane >> 4) * 8;
      int pcol = lane & 15;
#pragma unroll
      for (int nt = 0; nt < 2; nt++)
#pragma unroll
        for (int r = 0; r < 8; r++)
          Plds[wave][prow + r][nt * 16 + pcol] = f2bf(S[nt][r]);
    }
    asm volatile("s_wait_dscnt 0x0" ::: "memory");
    FragBF pf;
    {
      int prow = lane & 15;
      int pk   = (lane >> 4) * 8;
      pf.h[0] = *(const ushort8*)&Plds[wave][prow][pk];
      pf.h[1] = *(const ushort8*)&Plds[wave][prow][16 + pk];
    }

    // ---- O += P * V ----
#pragma unroll
    for (int n = 0; n < 4; n++) {
      int dcol = n * 16 + (lane & 15);
      int kgv  = (lane >> 4) * 8;
      FragBF bf_;
      bf_.h[0] = *(const ushort8*)&Vtld[buf][dcol][kgv];
      bf_.h[1] = *(const ushort8*)&Vtld[buf][dcol][kgv + 16];
      Of[n] = __builtin_amdgcn_wmma_f32_16x16x32_bf16(
          false, pf.v, false, bf_.v, (short)0, Of[n], false, false);
    }
  }

  const int rbase = r0 + (lane >> 4) * 8;
#pragma unroll
  for (int r = 0; r < 8; r++) {
    int row = rbase + r;
    if (row < qRows) {
      float inv = 1.0f / lrun[r];
#pragma unroll
      for (int n = 0; n < 4; n++)
        o[(size_t)row * ldo + n * 16 + (lane & 15)] = Of[n][r] * inv;
    }
  }
}

// =====================================================================
// ProbSparse sampled scores (f32 Q/K): one wave per (b,h,l)
// =====================================================================
__global__ __launch_bounds__(128)
void prob_scores_kernel(const float* __restrict__ Q, const float* __restrict__ K,
                        const int* __restrict__ idx, float* __restrict__ Mout)
{
  int gw   = (int)blockIdx.x * 4 + (threadIdx.x >> 5);
  int lane = threadIdx.x & 31;
  int l  = gw % SEQ_LQ;
  int bh = gw / SEQ_LQ;
  int h  = bh % NHEAD;
  int b  = bh / NHEAD;

  const float* qp = Q + (size_t)(b * SEQ_LQ + l) * DMODEL + h * DHEAD;
  float q0 = qp[lane], q1 = qp[lane + 32];
  float mx = -3.0e38f, sm = 0.0f;
  for (int u = 0; u < USEL; u++) {
    int ks = idx[l * USEL + u];
    const float* kp = K + (size_t)(b * SEQ_LK + ks) * DMODEL + h * DHEAD;
    float p = q0 * kp[lane] + q1 * kp[lane + 32];
#pragma unroll
    for (int off = 16; off; off >>= 1) p += __shfl_xor(p, off, 32);
    mx = fmaxf(mx, p);
    sm += p;
  }
  if (lane == 0) Mout[gw] = mx - sm / (float)SEQ_LK;
}

// =====================================================================
// Iterative top-U selection per (b,h)
// =====================================================================
__global__ __launch_bounds__(256)
void topk_kernel(const float* __restrict__ Mv, int* __restrict__ top)
{
  __shared__ float vals[SEQ_LQ];
  __shared__ float rv[256];
  __shared__ int   ri[256];
  int bh = blockIdx.x, tid = threadIdx.x;
  for (int i = tid; i < SEQ_LQ; i += 256) vals[i] = Mv[(size_t)bh * SEQ_LQ + i];
  __syncthreads();
  for (int t = 0; t < USEL; t++) {
    float best = -3.0e38f; int bi = 0;
    for (int i = tid; i < SEQ_LQ; i += 256) {
      float x = vals[i];
      if (x > best) { best = x; bi = i; }
    }
    rv[tid] = best; ri[tid] = bi;
    __syncthreads();
    for (int s = 128; s > 0; s >>= 1) {
      if (tid < s && rv[tid + s] > rv[tid]) { rv[tid] = rv[tid + s]; ri[tid] = ri[tid + s]; }
      __syncthreads();
    }
    if (tid == 0) { top[bh * USEL + t] = ri[0]; vals[ri[0]] = -3.0e38f; }
    __syncthreads();
  }
}

// gather reduced queries into padded (BH, 64, 64) bf16 buffer (zeros past USEL)
__global__ void gather_q_kernel(const float* __restrict__ Q, const int* __restrict__ top,
                                u16* __restrict__ qred)
{
  int bh = blockIdx.x, h = bh % NHEAD, b = bh / NHEAD, d = threadIdx.x;  // 64 threads
  for (int u = 0; u < 64; u++) {
    u16 val = 0;
    if (u < USEL) {
      int l = top[bh * USEL + u];
      val = f2bf(Q[(size_t)(b * SEQ_LQ + l) * DMODEL + h * DHEAD + d]);
    }
    qred[((size_t)bh * 64 + u) * 64 + d] = val;
  }
}

// mean of V over keys, per (b,h,d)
__global__ void vmean_kernel(const float* __restrict__ V, float* __restrict__ vm)
{
  int bh = blockIdx.x, h = bh % NHEAD, b = bh / NHEAD, d = threadIdx.x;  // 64 threads
  float s = 0.0f;
  for (int l = 0; l < SEQ_LK; l++)
    s += V[(size_t)(b * SEQ_LK + l) * DMODEL + h * DHEAD + d];
  vm[bh * DHEAD + d] = s * (1.0f / SEQ_LK);
}

// ctx = broadcast(v-mean) in merged (B, L, DMODEL) layout
__global__ void ctx_broadcast_kernel(const float* __restrict__ vm, float* __restrict__ ctx)
{
  int i = (int)blockIdx.x * 256 + threadIdx.x;
  int d  = i & (DMODEL - 1);
  int bl = i / DMODEL;
  int b  = bl / SEQ_LQ;
  int h  = d / DHEAD;
  ctx[i] = vm[(b * NHEAD + h) * DHEAD + (d & (DHEAD - 1))];
}

// scatter upd rows into ctx at top indices
__global__ void ctx_scatter_kernel(const float* __restrict__ upd, const int* __restrict__ top,
                                   float* __restrict__ ctx)
{
  int u = blockIdx.x, bh = blockIdx.y, d = threadIdx.x;  // 64 threads
  int h = bh % NHEAD, b = bh / NHEAD;
  int l = top[bh * USEL + u];
  ctx[(size_t)(b * SEQ_LQ + l) * DMODEL + h * DHEAD + d] =
      upd[((size_t)bh * 64 + u) * 64 + d];
}

// =====================================================================
// Fused (residual+)LayerNorm, optional dual f32/bf16 output
// =====================================================================
__global__ __launch_bounds__(128)
void layernorm_kernel(const float* __restrict__ X, const float* __restrict__ R,
                      const float* __restrict__ g, const float* __restrict__ bta,
                      float* __restrict__ out, u16* __restrict__ outBf,
                      int useRes, int mode)
{
  __shared__ float red[4];
  int row = blockIdx.x, tid = threadIdx.x;
  const float* x  = X + (size_t)row * DMODEL;
  const float* rp = R + (size_t)row * DMODEL;
  float loc[4], s = 0.0f;
#pragma unroll
  for (int j = 0; j < 4; j++) {
    int i = tid + j * 128;
    float vv = x[i];
    if (useRes) vv += rp[i];
    loc[j] = vv; s += vv;
  }
#pragma unroll
  for (int off = 16; off; off >>= 1) s += __shfl_xor(s, off, 32);
  if ((tid & 31) == 0) red[tid >> 5] = s;
  __syncthreads();
  float mean = (red[0] + red[1] + red[2] + red[3]) * (1.0f / DMODEL);
  float var = 0.0f;
#pragma unroll
  for (int j = 0; j < 4; j++) { float d2 = loc[j] - mean; var += d2 * d2; }
#pragma unroll
  for (int off = 16; off; off >>= 1) var += __shfl_xor(var, off, 32);
  __syncthreads();
  if ((tid & 31) == 0) red[tid >> 5] = var;
  __syncthreads();
  var = (red[0] + red[1] + red[2] + red[3]) * (1.0f / DMODEL);
  float inv = rsqrtf(var + 1e-5f);
#pragma unroll
  for (int j = 0; j < 4; j++) {
    int i = tid + j * 128;
    float val = (loc[j] - mean) * inv * g[i] + bta[i];
    size_t o = (size_t)row * DMODEL + i;
    if (mode & 1) out[o] = val;
    if (mode & 2) outBf[o] = f2bf(val);
  }
}

// elementwise f32 -> bf16
__global__ void cvt_bf16_kernel(const float* __restrict__ in, u16* __restrict__ out, int n)
{
  int i = (int)blockIdx.x * 256 + threadIdx.x;
  if (i < n) out[i] = f2bf(in[i]);
}

// W[K][N] f32 -> Wt[N][K] bf16 (transpose + convert, write-coalesced)
__global__ void wt_cvt_kernel(const float* __restrict__ W, u16* __restrict__ Wt, int K, int N)
{
  int i = (int)blockIdx.x * 256 + threadIdx.x;
  if (i >= N * K) return;
  int k = i % K, n = i / K;
  Wt[i] = f2bf(W[(size_t)k * N + n]);
}

// =====================================================================
extern "C" void kernel_launch(void* const* d_in, const int* in_sizes, int n_in,
                              void* d_out, int out_size, void* d_ws, size_t ws_size,
                              hipStream_t stream)
{
  (void)in_sizes; (void)n_in; (void)out_size; (void)ws_size;
  const float* tgt    = (const float*)d_in[0];
  const float* memory = (const float*)d_in[1];
  const int*   idxs   = (const int*)d_in[2];
  const float* wq1 = (const float*)d_in[3];  const float* bq1 = (const float*)d_in[4];
  const float* wk1 = (const float*)d_in[5];  const float* bk1 = (const float*)d_in[6];
  const float* wv1 = (const float*)d_in[7];  const float* bv1 = (const float*)d_in[8];
  const float* wo1 = (const float*)d_in[9];  const float* bo1 = (const float*)d_in[10];
  const float* wq2 = (const float*)d_in[11]; const float* bq2 = (const float*)d_in[12];
  const float* wk2 = (const float*)d_in[13]; const float* bk2 = (const float*)d_in[14];
  const float* wv2 = (const float*)d_in[15]; const float* bv2 = (const float*)d_in[16];
  const float* wo2 = (const float*)d_in[17]; const float* bo2 = (const float*)d_in[18];
  const float* c1w = (const float*)d_in[19]; const float* c1b = (const float*)d_in[20];
  const float* c2w = (const float*)d_in[21]; const float* c2b = (const float*)d_in[22];
  const float* ln1g = (const float*)d_in[23]; const float* ln1b = (const float*)d_in[24];
  const float* ln2g = (const float*)d_in[25]; const float* ln2b = (const float*)d_in[26];
  const float* ln3g = (const float*)d_in[27]; const float* ln3b = (const float*)d_in[28];
  float* outp = (float*)d_out;

  char* ws = (char*)d_ws;
  const size_t MB = (size_t)1 << 20;
  // f32 buffers
  float* Qb   = (float*)(ws +   0 * MB);  // 16MB
  float* Kb   = (float*)(ws +  16 * MB);  // 16MB
  float* Vb   = (float*)(ws +  32 * MB);  // 16MB
  float* ATTb = (float*)(ws +  48 * MB);  // 16MB
  float* Xb   = (float*)(ws +  64 * MB);  // 16MB (LN2 f32 out for residual)
  float* Tb   = (float*)(ws +  80 * MB);  // 16MB (pre-LN projections)
  // bf16 buffers
  u16* Qbf = (u16*)(ws +  96 * MB);  // 8MB
  u16* Kbf = (u16*)(ws + 104 * MB);  // 8MB
  u16* Vbf = (u16*)(ws + 112 * MB);  // 8MB
  u16* Abf = (u16*)(ws + 120 * MB);  // 8MB  (generic bf16 GEMM A)
  u16* Xbf = (u16*)(ws + 128 * MB);  // 8MB  (LN bf16 out)
  u16* Hbf = (u16*)(ws + 136 * MB);  // 32MB (FFN hidden, bf16 only)
  // bf16 transposed weights
  u16* wq1t = (u16*)(ws + 168 * MB); u16* wk1t = (u16*)(ws + 168 * MB + 512 * 1024);
  u16* wv1t = (u16*)(ws + 169 * MB); u16* wo1t = (u16*)(ws + 169 * MB + 512 * 1024);
  u16* wq2t = (u16*)(ws + 170 * MB); u16* wk2t = (u16*)(ws + 170 * MB + 512 * 1024);
  u16* wv2t = (u16*)(ws + 171 * MB); u16* wo2t = (u16*)(ws + 171 * MB + 512 * 1024);
  u16* c1t  = (u16*)(ws + 172 * MB); // 2MB
  u16* c2t  = (u16*)(ws + 174 * MB); // 2MB
  // small buffers
  float* Mbuf   = (float*)(ws + 176 * MB);           // 256KB
  int*   TOPb   = (int*)  (ws + 176 * MB + 512 * 1024);
  u16*   QREDbf = (u16*)  (ws + 177 * MB);           // 512KB
  float* UPDb   = (float*)(ws + 178 * MB);           // 1MB
  float* VMb    = (float*)(ws + 179 * MB);           // 16KB

  const int Mrows = BATCH * SEQ_LQ;  // 8192
  const int BH = BATCH * NHEAD;      // 64
  const int NELEM = Mrows * DMODEL;  // 4M
  dim3 blk128(128), blk256(256), blk64(64);
  dim3 g512(Mrows / 64, DMODEL / 64);
  dim3 g2048(Mrows / 64, DFF / 64);
  dim3 gElem(NELEM / 256);
  const float scale = 0.125f;  // 1/sqrt(64)

  // ---- one-time bf16 weight transposes ----
  {
    dim3 gw((DMODEL * DMODEL + 255) / 256);
    wt_cvt_kernel<<<gw, blk256, 0, stream>>>(wq1, wq1t, DMODEL, DMODEL);
    wt_cvt_kernel<<<gw, blk256, 0, stream>>>(wk1, wk1t, DMODEL, DMODEL);
    wt_cvt_kernel<<<gw, blk256, 0, stream>>>(wv1, wv1t, DMODEL, DMODEL);
    wt_cvt_kernel<<<gw, blk256, 0, stream>>>(wo1, wo1t, DMODEL, DMODEL);
    wt_cvt_kernel<<<gw, blk256, 0, stream>>>(wq2, wq2t, DMODEL, DMODEL);
    wt_cvt_kernel<<<gw, blk256, 0, stream>>>(wk2, wk2t, DMODEL, DMODEL);
    wt_cvt_kernel<<<gw, blk256, 0, stream>>>(wv2, wv2t, DMODEL, DMODEL);
    wt_cvt_kernel<<<gw, blk256, 0, stream>>>(wo2, wo2t, DMODEL, DMODEL);
    dim3 gc((DMODEL * DFF + 255) / 256);
    wt_cvt_kernel<<<gc, blk256, 0, stream>>>(c1w, c1t, DMODEL, DFF);   // Wt[N=2048][K=512]
    wt_cvt_kernel<<<gc, blk256, 0, stream>>>(c2w, c2t, DFF, DMODEL);   // Wt[N=512][K=2048]
  }

  // ---- self-attention projections (A = bf16(tgt)) ----
  cvt_bf16_kernel<<<gElem, blk256, 0, stream>>>(tgt, Abf, NELEM);
  gemm_bf16_kernel<<<g512, blk128, 0, stream>>>(Abf, wq1t, bq1, Qb, Qbf, Mrows, DMODEL, DMODEL, 0, 3);
  gemm_bf16_kernel<<<g512, blk128, 0, stream>>>(Abf, wk1t, bk1, Kb, Kbf, Mrows, DMODEL, DMODEL, 0, 3);
  gemm_bf16_kernel<<<g512, blk128, 0, stream>>>(Abf, wv1t, bv1, Vb, Vbf, Mrows, DMODEL, DMODEL, 0, 3);

  // ---- ProbSparse selection ----
  prob_scores_kernel<<<dim3(BH * SEQ_LQ / 4), blk128, 0, stream>>>(Qb, Kb, idxs, Mbuf);
  topk_kernel<<<dim3(BH), blk256, 0, stream>>>(Mbuf, TOPb);
  gather_q_kernel<<<dim3(BH), blk64, 0, stream>>>(Qb, TOPb, QREDbf);

  // ---- attention on reduced queries (padded 64 rows, 35 valid) ----
  flash_attn_kernel<<<dim3(BH), blk128, 0, stream>>>(
      QREDbf, 64, (size_t)NHEAD * 64 * 64, (size_t)64 * 64,
      Kbf, DMODEL, (size_t)SEQ_LK * DMODEL, (size_t)DHEAD,
      Vbf,
      UPDb, 64, (size_t)NHEAD * 64 * 64, (size_t)64 * 64,
      USEL, SEQ_LK, scale, 1);

  // ---- ctx = broadcast(mean V); scatter top rows ----
  vmean_kernel<<<dim3(BH), blk64, 0, stream>>>(Vb, VMb);
  ctx_broadcast_kernel<<<gElem, blk256, 0, stream>>>(VMb, ATTb);
  ctx_scatter_kernel<<<dim3(USEL, BH), blk64, 0, stream>>>(UPDb, TOPb, ATTb);

  // ---- output proj + LN1 (residual with tgt; bf16 out feeds q2 proj) ----
  cvt_bf16_kernel<<<gElem, blk256, 0, stream>>>(ATTb, Abf, NELEM);
  gemm_bf16_kernel<<<g512, blk128, 0, stream>>>(Abf, wo1t, bo1, Tb, (u16*)0, Mrows, DMODEL, DMODEL, 0, 1);
  layernorm_kernel<<<dim3(Mrows), blk128, 0, stream>>>(Tb, tgt, ln1g, ln1b, Xb, Xbf, 1, 2);

  // ---- cross-attention projections (bf16 outputs only) ----
  gemm_bf16_kernel<<<g512, blk128, 0, stream>>>(Xbf, wq2t, bq2, (float*)0, Qbf, Mrows, DMODEL, DMODEL, 0, 2);
  cvt_bf16_kernel<<<gElem, blk256, 0, stream>>>(memory, Abf, NELEM);
  gemm_bf16_kernel<<<g512, blk128, 0, stream>>>(Abf, wk2t, bk2, (float*)0, Kbf, Mrows, DMODEL, DMODEL, 0, 2);
  gemm_bf16_kernel<<<g512, blk128, 0, stream>>>(Abf, wv2t, bv2, (float*)0, Vbf, Mrows, DMODEL, DMODEL, 0, 2);

  // ---- full cross attention (fused flash) ----
  flash_attn_kernel<<<dim3(BH * (SEQ_LQ / 64)), blk128, 0, stream>>>(
      Qbf, DMODEL, (size_t)SEQ_LQ * DMODEL, (size_t)DHEAD,
      Kbf, DMODEL, (size_t)SEQ_LK * DMODEL, (size_t)DHEAD,
      Vbf,
      ATTb, DMODEL, (size_t)SEQ_LQ * DMODEL, (size_t)DHEAD,
      SEQ_LQ, SEQ_LK, scale, SEQ_LQ / 64);

  // ---- output proj + LN2 (no residual; f32 + bf16 outs) ----
  cvt_bf16_kernel<<<gElem, blk256, 0, stream>>>(ATTb, Abf, NELEM);
  gemm_bf16_kernel<<<g512, blk128, 0, stream>>>(Abf, wo2t, bo2, Tb, (u16*)0, Mrows, DMODEL, DMODEL, 0, 1);
  layernorm_kernel<<<dim3(Mrows), blk128, 0, stream>>>(Tb, Tb, ln2g, ln2b, Xb, Xbf, 0, 3);

  // ---- FFN + LN3 (residual with Xb) ----
  gemm_bf16_kernel<<<g2048, blk128, 0, stream>>>(Xbf, c1t, c1b, (float*)0, Hbf, Mrows, DFF, DMODEL, 1, 2);
  gemm_bf16_kernel<<<g512, blk128, 0, stream>>>(Hbf, c2t, c2b, Tb, (u16*)0, Mrows, DMODEL, DFF, 0, 1);
  layernorm_kernel<<<dim3(Mrows), blk128, 0, stream>>>(Tb, Xb, ln3g, ln3b, outp, (u16*)0, 1, 1);
}
